// RRWPEncoding_12000138625152
// MI455X (gfx1250) — matrix operand order repaired and checked by reading, not compiled, not run on hardware
//
#include <hip/hip_runtime.h>
#include <hip/hip_bf16.h>

// RRWP diagonal positional encoding, CDNA5/gfx1250.
// One workgroup (1024 thr = 32 wave32) per batch matrix.
// LDS: M_k (bf16, padded pitch, A-operand) + P^T (bf16, padded, B-operand).
// Chain of 6 in-LDS matmuls via v_wmma_f32_16x16x32_bf16, f32 accumulate.
// Pitch 264 (132 dwords, == 4 mod 64 banks) kills the row->bank aliasing that
// a 256-element pitch (128 dwords == 0 mod 64) would cause on fragment loads.

typedef __attribute__((ext_vector_type(16))) __bf16   v16bf;
typedef __attribute__((ext_vector_type(8)))  float    v8f;
typedef __attribute__((ext_vector_type(4)))  uint32_t v4u;
typedef __attribute__((ext_vector_type(2)))  uint32_t v2u;

#define RN 256          // N
#define RWL 8           // walk length
#define RTHRESH 0.3f
#define PITCH 264       // bf16 elements per LDS row (256 + 8 pad)

// LDS partition offsets (bytes)
#define OFF_M   0                        // 256*264*2 = 135168
#define OFF_PT  135168                   // 256*264*2 = 135168
#define OFF_DI  270336                   // 256 f32   = 1024
#define OFF_RED 271360                   // 1024 f32  = 4096
#define SMEM_BYTES 275456

union FragBF { v16bf v; v4u q[2]; };
union PackBF { __bf16 h[4]; v2u u2; };

__global__ __launch_bounds__(1024, 1)
void rrwp_wmma_kernel(const float* __restrict__ Ain,
                      float* __restrict__ Out)
{
    extern __shared__ char smem_raw[];
    __bf16* sM   = (__bf16*)(smem_raw + OFF_M);
    __bf16* sPt  = (__bf16*)(smem_raw + OFF_PT);
    float*  sDi  = (float*)(smem_raw + OFF_DI);
    float*  sRed = (float*)(smem_raw + OFF_RED);

    const int t = threadIdx.x;                  // 0..1023
    const int b = blockIdx.x;                   // batch
    const float* A  = Ain + (size_t)b * RN * RN;
    float*       oB = Out + (size_t)b * RN * RWL;

    // ---- Phase 0: thresholded row sums -> deg_inv (float4 coalesced) ----
    {
        const int row = t >> 2, part = t & 3;   // 4 threads per row
        const float4* Ar4 = (const float4*)(A + row * RN);
        float s = 0.f;
        #pragma unroll 4
        for (int i = 0; i < 16; ++i) {
            float4 v = Ar4[part + 4 * i];       // 64B per 4 lanes, coalesced
            s += (v.x > RTHRESH ? v.x : 0.f) + (v.y > RTHRESH ? v.y : 0.f)
               + (v.z > RTHRESH ? v.z : 0.f) + (v.w > RTHRESH ? v.w : 0.f);
        }
        sRed[t] = s;
        __syncthreads();
        if (part == 0) {
            float d = sRed[t] + sRed[t + 1] + sRed[t + 2] + sRed[t + 3];
            sDi[row] = (d > 0.f) ? (1.f / d) : 0.f;
        }
        __syncthreads();
    }

    // ---- Phase 1: build P (packed b64 stores) and P^T; emit cols 0 & 1 ----
    for (int e = t * 4; e < RN * RN; e += 4096) {
        int r = e >> 8, c = e & 255;            // c multiple of 4
        float4 a4 = *(const float4*)(A + e);
        float di = sDi[r];
        float p0 = (a4.x > RTHRESH ? a4.x : 0.f) * di;
        float p1 = (a4.y > RTHRESH ? a4.y : 0.f) * di;
        float p2 = (a4.z > RTHRESH ? a4.z : 0.f) * di;
        float p3 = (a4.w > RTHRESH ? a4.w : 0.f) * di;
        PackBF pk;
        pk.h[0] = (__bf16)p0; pk.h[1] = (__bf16)p1;
        pk.h[2] = (__bf16)p2; pk.h[3] = (__bf16)p3;
        *(v2u*)(sM + r * PITCH + c) = pk.u2;    // 8B aligned (528*r + 2c)
        sPt[(c + 0) * PITCH + r] = pk.h[0];     // transpose scatter
        sPt[(c + 1) * PITCH + r] = pk.h[1];
        sPt[(c + 2) * PITCH + r] = pk.h[2];
        sPt[(c + 3) * PITCH + r] = pk.h[3];
        if (r >= c && r < c + 4) {              // diagonal element in this quad
            float pd = (r - c == 0) ? p0 : (r - c == 1) ? p1
                     : (r - c == 2) ? p2 : p3;
            oB[r * RWL + 0] = 1.0f;             // identity term
            oB[r * RWL + 1] = pd;               // diag(P), f32 precision
        }
    }
    __syncthreads();

    // ---- Phase 2: chained powers M_k = M_{k-1} @ P, diag extract ----
    // 32 waves, each owns a 2x4 block of 16x16 tiles (16x16 tile grid).
    const int wave = t >> 5;
    const int lane = t & 31;
    const int half = lane >> 4;        // half-wave selector
    const int mn   = lane & 15;        // A row within tile / B-C column
    const int ti0  = (wave >> 2) * 2;  // first tile row of block
    const int tj0  = (wave & 3) * 4;   // first tile col of block

    for (int k = 2; k < RWL; ++k) {
        v8f acc[2][4];
        #pragma unroll
        for (int ri = 0; ri < 2; ++ri)
            #pragma unroll
            for (int jc = 0; jc < 4; ++jc)
                acc[ri][jc] = (v8f){0.f,0.f,0.f,0.f,0.f,0.f,0.f,0.f};

        for (int kk = 0; kk < 8; ++kk) {       // K steps of 32
            // A fragments: lane holds K = half*8 + {0..7, 16..23}, two b128s
            FragBF a[2];
            #pragma unroll
            for (int ri = 0; ri < 2; ++ri) {
                const uint32_t* rowM = (const uint32_t*)
                    (sM + ((ti0 + ri) * 16 + mn) * PITCH) + kk * 16 + half * 4;
                a[ri].q[0] = *(const v4u*)(rowM);
                a[ri].q[1] = *(const v4u*)(rowM + 8);
            }
            #pragma unroll
            for (int jc = 0; jc < 4; ++jc) {
                // B fragment: lane holds K = half*8 + {0..7, 16..23}, two b128s
                FragBF bf;
                const uint32_t* rowP = (const uint32_t*)
                    (sPt + ((tj0 + jc) * 16 + mn) * PITCH) + kk * 16 + half * 4;
                bf.q[0] = *(const v4u*)(rowP);
                bf.q[1] = *(const v4u*)(rowP + 8);
                #pragma unroll
                for (int ri = 0; ri < 2; ++ri)
                    acc[ri][jc] = __builtin_amdgcn_wmma_f32_16x16x32_bf16(
                        false, a[ri].v, false, bf.v,
                        (short)0, acc[ri][jc], false, false);
            }
        }

        __syncthreads();   // all reads of M_{k-1} complete before overwrite

        #pragma unroll
        for (int ri = 0; ri < 2; ++ri) {
            int ti = ti0 + ri;
            #pragma unroll
            for (int jc = 0; jc < 4; ++jc) {
                int tj = tj0 + jc;
                #pragma unroll
                for (int r = 0; r < 8; ++r) {
                    int gm = ti * 16 + r + half * 8;  // C/D layout: M=r+8*half
                    int gn = tj * 16 + mn;            // N = lane%16
                    sM[gm * PITCH + gn] = (__bf16)acc[ri][jc][r];
                }
                if (ti == tj) {  // diagonal tile -> diag(P^k) in f32
                    if (half == 0 && mn < 8)
                        oB[(ti * 16 + mn) * RWL + k] = acc[ri][jc][mn];
                    if (half == 1 && mn >= 8)
                        oB[(ti * 16 + mn) * RWL + k] = acc[ri][jc][mn - 8];
                }
            }
        }
        __syncthreads();   // M_k visible before next iteration reads it
    }
}

extern "C" void kernel_launch(void* const* d_in, const int* in_sizes, int n_in,
                              void* d_out, int out_size, void* d_ws, size_t ws_size,
                              hipStream_t stream) {
    const float* A = (const float*)d_in[0];
    float* out = (float*)d_out;
    const int B = in_sizes[0] / (RN * RN);     // 256
    rrwp_wmma_kernel<<<B, 1024, SMEM_BYTES, stream>>>(A, out);
}
